// ChemiNet_4973572129015
// MI455X (gfx1250) — matrix operand order repaired and checked by reading, not compiled, not run on hardware
//
#include <hip/hip_runtime.h>
#include <hip/hip_bf16.h>

#define F_ATOM 75
#define F_BOND 12
#define OUTC   10
#define JTOT   (F_ATOM * OUTC)   // 750
#define NT     47                // ceil(750/16) column tiles

typedef __attribute__((ext_vector_type(16))) __bf16 v16bf;
typedef __attribute__((ext_vector_type(8)))  float  v8f;
typedef __attribute__((ext_vector_type(4)))  float  v4f;

// ---------------------------------------------------------------------------
// Pack [lin_w | lin_b] into WMMA B fragments (bf16), ISA layout:
// B is 32x16 (KxN). half h = lane/16: elem t<8 -> K=8h+t, elem t>=8 ->
// K=16+8h+(t-8).  B[k][n] with j = 16*tile + n:
//   k <  12 : lin_w[j][k]
//   k == 12 : lin_b[j]          (bias folded in as a rank-1 K-term)
//   else    : 0
// Buffer layout: [tile][lane][16 bf16] -> one contiguous v16bf per lane.
// ---------------------------------------------------------------------------
__global__ void prep_b_kernel(const float* __restrict__ lin_w,
                              const float* __restrict__ lin_b,
                              unsigned short* __restrict__ Bbuf_raw) {
    __bf16* Bbuf = (__bf16*)Bbuf_raw;
    int idx = blockIdx.x * blockDim.x + threadIdx.x;
    if (idx >= NT * 32 * 16) return;
    int tile = idx >> 9;          // /512
    int lane = (idx >> 4) & 31;
    int t    = idx & 15;
    int laneHi = lane >> 4;
    int n      = lane & 15;
    int j = tile * 16 + n;
    int k = (t < 8) ? (8 * laneHi + t) : (16 + 8 * laneHi + (t - 8));
    float v = 0.f;
    if (j < JTOT) {
        if (k < F_BOND)       v = lin_w[j * F_BOND + k];
        else if (k == F_BOND) v = lin_b[j];
    }
    Bbuf[idx] = (__bf16)v;
}

__global__ void zero_kernel(float* __restrict__ p, int n) {
    int i = blockIdx.x * blockDim.x + threadIdx.x;
    if (i < n) p[i] = 0.f;
}

// ---------------------------------------------------------------------------
// Fused edge kernel: per 16-edge tile (one wave each):
//   W[e][j] = relu( [edge_attr[e],1] @ [lin_w^T;lin_b] )   via WMMA bf16, C=0
//   msg[e][o] += W[e][j] * x[src[e]][j/10]   (o = j%10)    via LDS f32 atomics
//   agg[dst[e]][o] += msg[e][o]                            via global atomics
// x rows are gathered (transposed) into LDS with GLOBAL_LOAD_ASYNC_TO_LDS.
// ---------------------------------------------------------------------------
__global__ __launch_bounds__(256) void edge_kernel(
    const float* __restrict__ x,
    const int*   __restrict__ edge_index,   // [2,E] flat: src then dst
    const float* __restrict__ edge_attr,    // [E,12]
    const unsigned short* __restrict__ Bbuf_raw,
    float* __restrict__ agg,                // [N,10], pre-zeroed
    int E)
{
    __shared__ __align__(16) float sX[8][76][16];  // [wave][i][e] transposed
    __shared__ float sMsg[8][160];                 // per-edge msg accumulators
    __shared__ int   sSrc[8][16];
    __shared__ int   sDst[8][16];

    const int tid    = threadIdx.x;
    const int lane   = tid & 31;
    const int wv     = tid >> 5;        // wave id in block (0..7)
    const int laneHi = lane >> 4;
    const int ln     = lane & 15;

    const int tile = blockIdx.x * 8 + wv;
    const int e0   = tile * 16;

    // edge meta (invalid edges -> node 0; their rows are masked at scatter)
    {
        int er = e0 + ln;
        if (laneHi == 0) sSrc[wv][ln] = (er < E) ? edge_index[er]     : 0;
        else             sDst[wv][ln] = (er < E) ? edge_index[E + er] : 0;
    }
    for (int i = lane; i < 160; i += 32) sMsg[wv][i] = 0.f;

    // async-gather the 16 source-node rows of x into LDS, transposed to [i][e]
    for (int c = lane; c < 16 * F_ATOM; c += 32) {
        int e = c / F_ATOM, i = c - e * F_ATOM;
        const float* gp = x + (size_t)sSrc[wv][e] * F_ATOM + i;
        unsigned lo = (unsigned)(uintptr_t)&sX[wv][i][e];   // LDS byte offset
        asm volatile("global_load_async_to_lds_b32 %0, %1, off"
                     :: "v"(lo), "v"(gp) : "memory");
    }

    // build A fragment (16x32 bf16, ISA layout): K=0..11 = edge_attr,
    // K=12 = 1.0 (bias term), rest zero.  Invalid edges -> all-zero row
    // (W row = relu(0) = 0, contributes nothing).
    v16bf afrag;
    #pragma unroll
    for (int t = 0; t < 16; ++t) afrag[t] = (__bf16)0.f;
    {
        int er = e0 + ln;
        if (er < E) {
            const float* ea = edge_attr + er * F_BOND;
            if (laneHi == 0) {
                #pragma unroll
                for (int t = 0; t < 8; ++t) afrag[t] = (__bf16)ea[t];      // K=0..7
            } else {
                #pragma unroll
                for (int t = 0; t < 4; ++t) afrag[t] = (__bf16)ea[8 + t];  // K=8..11
                afrag[4] = (__bf16)1.0f;                                   // K=12
            }
        }
    }

    asm volatile("s_wait_asynccnt 0x0" ::: "memory");   // x rows landed in LDS
    __syncthreads();

    const v16bf* Bp = (const v16bf*)Bbuf_raw;

    // software-pipelined B fragments: issue t+1's load before consuming t's
    v16bf bnext = Bp[lane];
    for (int t = 0; t < NT; ++t) {
        v16bf bfrag = bnext;
        if (t + 1 < NT) bnext = Bp[(t + 1) * 32 + lane];

        v8f c = {};   // inline-0 C operand
        c = __builtin_amdgcn_wmma_f32_16x16x32_bf16(
                false, afrag, false, bfrag, (short)0, c, false, false);

        // D layout: lane holds column n=ln; element r -> edge row r + 8*laneHi
        int j = t * 16 + ln;
        if (j < JTOT) {
            int i = j / OUTC;
            int o = j - i * OUTC;
            // lane's 8 edges are contiguous in transposed sX: 2x ds_load_b128
            const v4f* xp = (const v4f*)&sX[wv][i][laneHi * 8];
            v4f x0 = xp[0];
            v4f x1 = xp[1];
            #pragma unroll
            for (int r = 0; r < 8; ++r) {
                float xv   = (r < 4) ? x0[r] : x1[r - 4];
                float wval = fmaxf(c[r], 0.0f);          // relu(W + bias)
                atomicAdd(&sMsg[wv][(r + 8 * laneHi) * OUTC + o], wval * xv);
            }
        }
    }
    __syncthreads();

    // scatter-add msg -> agg[dst]
    #pragma unroll
    for (int p = 0; p < 5; ++p) {
        int idx = lane + 32 * p;            // 160 (e,o) pairs
        int e = idx / OUTC, o = idx - e * OUTC;
        int er = e0 + e;
        if (er < E) atomicAdd(&agg[sDst[wv][e] * OUTC + o], sMsg[wv][idx]);
    }
}

// h = agg + x @ root_w + conv_b ; pooled[batch[n]] += h
__global__ void node_kernel(const float* __restrict__ x,
                            const float* __restrict__ agg,
                            const float* __restrict__ root_w,  // [75,10]
                            const float* __restrict__ conv_b,  // [10]
                            const int*   __restrict__ batch,
                            float* __restrict__ pooled,        // [G,10], zeroed
                            int N)
{
    int idx = blockIdx.x * blockDim.x + threadIdx.x;
    if (idx >= N * OUTC) return;
    int n = idx / OUTC, o = idx - n * OUTC;
    float acc = agg[idx] + conv_b[o];
    const float* xr = x + n * F_ATOM;
    #pragma unroll 5
    for (int i = 0; i < F_ATOM; ++i) acc += xr[i] * root_w[i * OUTC + o];
    atomicAdd(&pooled[batch[n] * OUTC + o], acc);
}

// out[g] = pooled[g] . out_w + out_b
__global__ void final_kernel(const float* __restrict__ pooled,
                             const float* __restrict__ out_w,  // [1,10]
                             const float* __restrict__ out_b,  // [1]
                             float* __restrict__ out, int G)
{
    int g = blockIdx.x * blockDim.x + threadIdx.x;
    if (g >= G) return;
    float a = out_b[0];
    #pragma unroll
    for (int o = 0; o < OUTC; ++o) a += pooled[g * OUTC + o] * out_w[o];
    out[g] = a;
}

extern "C" void kernel_launch(void* const* d_in, const int* in_sizes, int n_in,
                              void* d_out, int out_size, void* d_ws, size_t ws_size,
                              hipStream_t stream) {
    const float* x         = (const float*)d_in[0];
    const int*   edge_index= (const int*)  d_in[1];
    const float* edge_attr = (const float*)d_in[2];
    const int*   batch     = (const int*)  d_in[3];
    const float* lin_w     = (const float*)d_in[4];
    const float* lin_b     = (const float*)d_in[5];
    const float* root_w    = (const float*)d_in[6];
    const float* conv_b    = (const float*)d_in[7];
    const float* out_w     = (const float*)d_in[8];
    const float* out_b     = (const float*)d_in[9];
    float* out = (float*)d_out;

    const int N = in_sizes[0] / F_ATOM;
    const int E = in_sizes[2] / F_BOND;
    const int G = out_size;               // output is [G,1]

    // workspace layout
    char* ws = (char*)d_ws;
    unsigned short* Bbuf = (unsigned short*)ws;        // 47*32*16 bf16 = 48128 B
    float* agg    = (float*)(ws + 49152);              // [N,10]
    float* pooled = agg + (size_t)N * OUTC;            // [G,10] (contiguous)

    // 1) pack [lin_w | lin_b] into WMMA B fragments
    {
        int total = NT * 32 * 16;
        prep_b_kernel<<<(total + 255) / 256, 256, 0, stream>>>(lin_w, lin_b, Bbuf);
    }
    // 2) zero agg + pooled (contiguous)
    {
        int zn = N * OUTC + G * OUTC;
        zero_kernel<<<(zn + 255) / 256, 256, 0, stream>>>(agg, zn);
    }
    // 3) fused edge MLP (WMMA) + per-edge contraction + scatter
    {
        int tiles  = (E + 15) / 16;
        int blocks = (tiles + 7) / 8;     // 8 waves (tiles) per 256-thread block
        edge_kernel<<<blocks, 256, 0, stream>>>(x, edge_index, edge_attr,
                                                Bbuf, agg, E);
    }
    // 4) root transform + graph pooling
    {
        int total = N * OUTC;
        node_kernel<<<(total + 255) / 256, 256, 0, stream>>>(x, agg, root_w,
                                                             conv_b, batch, pooled, N);
    }
    // 5) final linear [G,10] -> [G,1]
    final_kernel<<<(G + 255) / 256, 256, 0, stream>>>(pooled, out_w, out_b, out, G);
}
